// QuantumDenseLayer_18769007083811
// MI455X (gfx1250) — compile-verified, matrix operand order checked by reading, not run on hardware
//
#include <hip/hip_runtime.h>

#define N_QUBITS 12
#define DEPTH    4
#define DIM      4096      // 2^12 == M_IN == OUT_DIM
#define BATCH    2048

typedef _Float16 v16h __attribute__((ext_vector_type(16)));
typedef _Float16 v8h  __attribute__((ext_vector_type(8)));
typedef _Float16 v4h  __attribute__((ext_vector_type(4)));
typedef float    v8f  __attribute__((ext_vector_type(8)));

union V16 { v16h v; v8h h[2]; };

// ---------------------------------------------------------------- Phase 0
__global__ __launch_bounds__(256) void qdl_convert_f16(const float* __restrict__ x,
                                                       _Float16* __restrict__ a)
{
    const int i = (blockIdx.x * 256 + threadIdx.x) * 4;
    float4 v = *(const float4*)(x + i);
    v4h h;
    h[0] = (_Float16)v.x; h[1] = (_Float16)v.y;
    h[2] = (_Float16)v.z; h[3] = (_Float16)v.w;
    *(v4h*)(a + i) = h;
}

// ---------------------------------------------------------------- Phase 1
// One block per basis state j: evolve e_j through the circuit in LDS and
// write COLUMN j of the transposed operator Wt[n][k] (Wt[n][j] = U[n,j]),
// so the GEMM's B operand is K-contiguous and needs no in-loop transpose.
__global__ __launch_bounds__(256) void qdl_build_unitary(const float* __restrict__ w,
                                                         _Float16* __restrict__ Wrt,
                                                         _Float16* __restrict__ Wit)
{
    __shared__ float sr[DIM];
    __shared__ float si[DIM];
    __shared__ unsigned short perm[DIM];   // CNOT-ladder permutation (gather map)

    const int tid  = threadIdx.x;
    const int bidx = blockIdx.x;

    // init state + precompute permutation: s_after[k] = s_before[perm[k]].
    // Gates applied forward as (i asc, j asc); the gather composition applies
    // the index maps in reverse gate order. Qubit q lives at bit (11-q).
    for (int k = tid; k < DIM; k += 256) {
        sr[k] = (k == bidx) ? 1.0f : 0.0f;
        si[k] = 0.0f;
        int src = k;
        for (int i = N_QUBITS - 2; i >= 0; --i)
            for (int j = N_QUBITS - 1; j > i; --j) {
                const int pc = N_QUBITS - 1 - i;
                const int pt = N_QUBITS - 1 - j;
                src ^= ((src >> pc) & 1) << pt;   // CNOT(c=i, t=j) gather
            }
        perm[k] = (unsigned short)src;
    }

    for (int layer = 0; layer < DEPTH; ++layer) {
        // 12 single-qubit Rot(phi,theta,omega) gates
        for (int q = 0; q < N_QUBITS; ++q) {
            const float phi = w[(layer * N_QUBITS + q) * 3 + 0];
            const float th  = w[(layer * N_QUBITS + q) * 3 + 1];
            const float om  = w[(layer * N_QUBITS + q) * 3 + 2];
            float c, s, cA, sA, cB, sB;
            sincosf(th * 0.5f, &s, &c);
            sincosf((phi + om) * 0.5f, &sA, &cA);
            sincosf((phi - om) * 0.5f, &sB, &cB);
            // U00 = c e^{-iA}, U01 = -s e^{iB}, U10 = s e^{-iB}, U11 = c e^{iA}
            const float u00r =  c * cA, u00i = -c * sA;
            const float u01r = -s * cB, u01i = -s * sB;
            const float u10r =  s * cB, u10i = -s * sB;
            const float u11r =  c * cA, u11i =  c * sA;

            const int bit     = N_QUBITS - 1 - q;
            const int lowmask = (1 << bit) - 1;
            __syncthreads();
            for (int p = tid; p < DIM / 2; p += 256) {
                const int k0 = ((p >> bit) << (bit + 1)) | (p & lowmask);
                const int k1 = k0 | (1 << bit);
                const float a0r = sr[k0], a0i = si[k0];
                const float a1r = sr[k1], a1i = si[k1];
                sr[k0] = u00r * a0r - u00i * a0i + u01r * a1r - u01i * a1i;
                si[k0] = u00r * a0i + u00i * a0r + u01r * a1i + u01i * a1r;
                sr[k1] = u10r * a0r - u10i * a0i + u11r * a1r - u11i * a1i;
                si[k1] = u10r * a0i + u10i * a0r + u11r * a1i + u11i * a1r;
            }
        }
        // apply the CNOT ladder as one permutation
        __syncthreads();
        float tr[16], ti[16];
        #pragma unroll
        for (int e = 0; e < 16; ++e) {
            const int k = e * 256 + tid;
            const int sidx = perm[k];
            tr[e] = sr[sidx]; ti[e] = si[sidx];
        }
        __syncthreads();
        #pragma unroll
        for (int e = 0; e < 16; ++e) {
            const int k = e * 256 + tid;
            sr[k] = tr[e]; si[k] = ti[e];
        }
    }

    __syncthreads();
    // transposed store: column bidx of Wt (scattered b16; merged in L2 —
    // Wt is fully L2-resident, one-time cost)
    for (int n = tid; n < DIM; n += 256) {
        Wrt[(size_t)n * DIM + bidx] = (_Float16)sr[n];
        Wit[(size_t)n * DIM + bidx] = (_Float16)si[n];
    }
}

// ---------------------------------------------------------------- Phase 2
// Fused complex GEMM + |.|^2 : out[b,n] = (X@Wr)^2 + (X@Wi)^2
// B operand (Wt) is pre-transposed: staging is pure b128 traffic.
#define BM 128
#define BN 64
#define BK 32
#define LDA 40   // padded LDS strides (halves)
#define LDB 40

__global__ __launch_bounds__(256) void qdl_gemm_probs(const _Float16* __restrict__ A,
                                                      const _Float16* __restrict__ Wrt,
                                                      const _Float16* __restrict__ Wit,
                                                      float* __restrict__ out)
{
    __shared__ _Float16 As [BM * LDA];
    __shared__ _Float16 Brs[BN * LDB];   // [n][k]
    __shared__ _Float16 Bis[BN * LDB];

    const int tid  = threadIdx.x;
    const int lane = tid & 31;
    const int wave = tid >> 5;
    const int wm   = wave >> 1;          // 0..3 -> M offset wm*32
    const int wn   = wave & 1;           // 0..1 -> N offset wn*32
    const int tileM = blockIdx.y * BM;
    const int tileN = blockIdx.x * BN;

    const int rrow  = lane & 15;         // row (A) / col (B) within 16x16
    const int khalf = lane >> 4;         // ISA 16-bit layout: K half select
    const int kbase = khalf * 8;         // elems 0..7 -> K kbase.., 8..15 -> K kbase+16..

    // staging maps (hoisted out of the loop)
    const int arow = tid >> 1;           // A: 128 rows, 2 threads/row
    const int aseg = (tid & 1) * 16;
    const int nrow = tid >> 2;           // B: 64 rows, 4 threads/row
    const int nseg = (tid & 3) * 8;

    v8f accr[2][2] = {};
    v8f acci[2][2] = {};

    for (int k0 = 0; k0 < DIM; k0 += BK) {
        __syncthreads();
        {   // A tile 128x32
            const _Float16* g = A + (size_t)(tileM + arow) * DIM + k0 + aseg;
            *(v8h*)(As + arow * LDA + aseg)     = *(const v8h*)g;
            *(v8h*)(As + arow * LDA + aseg + 8) = *(const v8h*)(g + 8);
        }
        {   // Wt tiles 64(n) x 32(k), already K-contiguous
            const size_t goff = (size_t)(tileN + nrow) * DIM + k0 + nseg;
            *(v8h*)(Brs + nrow * LDB + nseg) = *(const v8h*)(Wrt + goff);
            *(v8h*)(Bis + nrow * LDB + nseg) = *(const v8h*)(Wit + goff);
        }
        __syncthreads();

        V16 af[2], bfr[2], bfi[2];
        #pragma unroll
        for (int sm = 0; sm < 2; ++sm) {
            const _Float16* p = As + (wm * 32 + sm * 16 + rrow) * LDA + kbase;
            af[sm].h[0] = *(const v8h*)p;
            af[sm].h[1] = *(const v8h*)(p + 16);
        }
        #pragma unroll
        for (int sn = 0; sn < 2; ++sn) {
            const _Float16* pr = Brs + (wn * 32 + sn * 16 + rrow) * LDB + kbase;
            const _Float16* pi = Bis + (wn * 32 + sn * 16 + rrow) * LDB + kbase;
            bfr[sn].h[0] = *(const v8h*)pr; bfr[sn].h[1] = *(const v8h*)(pr + 16);
            bfi[sn].h[0] = *(const v8h*)pi; bfi[sn].h[1] = *(const v8h*)(pi + 16);
        }
        #pragma unroll
        for (int sm = 0; sm < 2; ++sm)
            #pragma unroll
            for (int sn = 0; sn < 2; ++sn) {
                accr[sm][sn] = __builtin_amdgcn_wmma_f32_16x16x32_f16(
                    false, af[sm].v, false, bfr[sn].v, (short)0, accr[sm][sn], false, false);
                acci[sm][sn] = __builtin_amdgcn_wmma_f32_16x16x32_f16(
                    false, af[sm].v, false, bfi[sn].v, (short)0, acci[sm][sn], false, false);
            }
    }

    // C layout: VGPR r, lane l -> M = r + (l<16?0:8), N = l%16
    const int rowhi = khalf * 8;
    #pragma unroll
    for (int sm = 0; sm < 2; ++sm)
        #pragma unroll
        for (int sn = 0; sn < 2; ++sn)
            #pragma unroll
            for (int r = 0; r < 8; ++r) {
                const int row = tileM + wm * 32 + sm * 16 + r + rowhi;
                const int col = tileN + wn * 32 + sn * 16 + rrow;
                const float vr = accr[sm][sn][r];
                const float vi = acci[sm][sn][r];
                out[(size_t)row * DIM + col] = vr * vr + vi * vi;
            }
}

// ---------------------------------------------------------------- Phase 3
__global__ __launch_bounds__(256) void qdl_row_normalize(float* __restrict__ out)
{
    __shared__ float red[256];
    const int b = blockIdx.x;
    const int tid = threadIdx.x;
    float* row = out + (size_t)b * DIM;

    float vals[16];
    float s = 0.f;
    #pragma unroll
    for (int e = 0; e < 16; ++e) { vals[e] = row[e * 256 + tid]; s += vals[e]; }
    red[tid] = s;
    __syncthreads();
    for (int off = 128; off > 0; off >>= 1) {
        if (tid < off) red[tid] += red[tid + off];
        __syncthreads();
    }
    const float tot = red[0];
    if (tot > 1e-10f) {
        const float inv = 1.0f / tot;
        #pragma unroll
        for (int e = 0; e < 16; ++e) row[e * 256 + tid] = vals[e] * inv;
    } else {
        #pragma unroll
        for (int e = 0; e < 16; ++e) row[e * 256 + tid] = 1.0f / (float)DIM;
    }
}

// ---------------------------------------------------------------- launch
extern "C" void kernel_launch(void* const* d_in, const int* in_sizes, int n_in,
                              void* d_out, int out_size, void* d_ws, size_t ws_size,
                              hipStream_t stream)
{
    (void)in_sizes; (void)n_in; (void)out_size; (void)ws_size;
    const float* x = (const float*)d_in[0];           // (2048, 4096) f32
    const float* w = (const float*)d_in[1];           // (4, 12, 3)   f32
    float* out = (float*)d_out;                       // (2048, 4096) f32
    char* ws = (char*)d_ws;

    _Float16* A16 = (_Float16*)ws;                                        // 16 MB
    _Float16* Wrt = (_Float16*)(ws + (size_t)BATCH * DIM * sizeof(_Float16));
    _Float16* Wit = Wrt + (size_t)DIM * DIM;                              // +32 MB each

    qdl_convert_f16 <<<(BATCH * DIM / 4) / 256, 256, 0, stream>>>(x, A16);
    qdl_build_unitary<<<DIM, 256, 0, stream>>>(w, Wrt, Wit);
    qdl_gemm_probs  <<<dim3(DIM / BN, BATCH / BM), 256, 0, stream>>>(A16, Wrt, Wit, out);
    qdl_row_normalize<<<BATCH, 256, 0, stream>>>(out);
}
